// SelfAttention_39831526703400
// MI455X (gfx1250) — matrix operand order repaired and checked by reading, not compiled, not run on hardware
//
#include <hip/hip_runtime.h>
#include <hip/hip_bf16.h>

// Problem constants (from reference): N=32, TOT=32768, D=56, DK=64, C=8, H=5
#define H_   5
#define C_   8
#define D_   56
#define DK_  64
#define CD_  448            // C_*D_ : flattened K of the Q GEMM
#define NL_  131072         // N * L = 32*4096 rows
#define ROWS 32             // rows (n,l) per block -> two WMMA M tiles per wave
#define MT   2              // M tiles per wave
#define THREADS 160         // 5 waves, one head per wave
#define OUTW 280            // H_*D_
#define ARENA_PER_WAVE 7168 // max(32*64*2 [Q bf16], 32*56*4 [v f32]) bytes

typedef __bf16 v16bf __attribute__((ext_vector_type(16)));
typedef float  v8f   __attribute__((ext_vector_type(8)));
typedef unsigned int u32x4 __attribute__((ext_vector_type(4)));
typedef unsigned int u32x2 __attribute__((ext_vector_type(2)));

union BF4 { __bf16 h[4]; u32x2 u; };
union BF8 { __bf16 h[8]; u32x4 u; };

// ---- weight prep: Wq (H,C,D,DK) f32 -> Wq_t (H,DK,CD) bf16 (t-major, K contiguous per col)
__global__ __launch_bounds__(256) void prep_wq(const float* __restrict__ Wq,
                                               __bf16* __restrict__ wq_t) {
  int idx = blockIdx.x * 256 + threadIdx.x;
  if (idx >= H_ * DK_ * CD_) return;
  int h  = idx / (DK_ * CD_);
  int r  = idx - h * (DK_ * CD_);
  int t  = r / CD_;
  int cd = r - t * CD_;
  wq_t[idx] = (__bf16)Wq[(h * CD_ + cd) * DK_ + t];   // ((h*C+c)*D+d)*DK+t
}

// ---- weight prep: Wk (H,D,DK) f32 -> wk_b (H,64,64) bf16, zero-padded rows d>=56
__global__ __launch_bounds__(256) void prep_wk(const float* __restrict__ Wk,
                                               __bf16* __restrict__ wk_b) {
  int idx = blockIdx.x * 256 + threadIdx.x;
  if (idx >= H_ * 64 * 64) return;
  int h = idx >> 12;
  int r = idx & 4095;
  int d = r >> 6;
  int t = r & 63;
  wk_b[idx] = (d < D_) ? (__bf16)Wk[(h * D_ + d) * DK_ + t] : (__bf16)0.0f;
}

// ---- fused attention: one pass over x, bf16 WMMA GEMMs + fp32 softmax/att
__global__ __launch_bounds__(THREADS) void attn_fused(
    const float*  __restrict__ x,      // (NL, CD) f32
    const __bf16* __restrict__ wq_t,   // (H, DK, CD) bf16
    const __bf16* __restrict__ wk_b,   // (H, 64, 64) bf16
    float*        __restrict__ out) {  // (NL, OUTW) f32

  __shared__ __attribute__((aligned(16))) __bf16 xbf[ROWS * CD_];          // 28672 B
  // time-shared arena: per-wave Q(bf16)/v(f32) slices, then block-wide att staging
  __shared__ __attribute__((aligned(16))) char arena[H_ * ARENA_PER_WAVE]; // 35840 B

  const int tid  = threadIdx.x;
  const int wave = tid >> 5;     // == head index (0..4)
  const int lane = tid & 31;
  const long long g0 = (long long)blockIdx.x * ROWS;

  // ---- cooperative load of the 32x448 f32 x-tile, converted to bf16 in LDS
  const float4* xsrc = reinterpret_cast<const float4*>(x + (size_t)g0 * CD_);
  for (int i = tid; i < ROWS * CD_ / 4; i += THREADS) {
    float4 v = xsrc[i];
    BF4 b;
    b.h[0] = (__bf16)v.x; b.h[1] = (__bf16)v.y;
    b.h[2] = (__bf16)v.z; b.h[3] = (__bf16)v.w;
    *reinterpret_cast<u32x2*>(&xbf[i * 4]) = b.u;
  }
  __syncthreads();

  // WMMA lane mapping (16-bit A 16x32 layout / B 32x16 layout, wave32)
  const int rowA  = lane & 15;
  const int koffA = (lane >> 4) * 8;
  const int colB  = lane & 15;
  const int koffB = (lane >> 4) * 8;
  const int rbase = (lane >> 4) * 8;

  // ================= Q = xtile(32x448) @ Wq[h](448x64); B reused across M tiles =====
  v8f qacc[MT][4] = {};
  const __bf16* wqh = wq_t + (size_t)wave * DK_ * CD_;
#pragma unroll
  for (int ks = 0; ks < CD_ / 32; ++ks) {          // 14 K-steps of 32
    v16bf a[MT];
#pragma unroll
    for (int mt = 0; mt < MT; ++mt) {
      u32x4* ap = reinterpret_cast<u32x4*>(&a[mt]);
      const __bf16* asrc = &xbf[(mt * 16 + rowA) * CD_ + ks * 32 + koffA];
      ap[0] = *reinterpret_cast<const u32x4*>(asrc);
      ap[1] = *reinterpret_cast<const u32x4*>(asrc + 16);
    }
#pragma unroll
    for (int nt = 0; nt < 4; ++nt) {
      v16bf b;
      u32x4* bp = reinterpret_cast<u32x4*>(&b);
      const __bf16* bsrc = wqh + (size_t)(nt * 16 + colB) * CD_ + ks * 32 + koffB;
      bp[0] = *reinterpret_cast<const u32x4*>(bsrc);
      bp[1] = *reinterpret_cast<const u32x4*>(bsrc + 16);
#pragma unroll
      for (int mt = 0; mt < MT; ++mt)
        qacc[mt][nt] = __builtin_amdgcn_wmma_f32_16x16x32_bf16(
            false, a[mt], false, b, (short)0, qacc[mt][nt], false, false);
    }
  }

  // spill Q to per-wave arena slice as bf16 row-major (32x64)
  {
    __bf16* q = reinterpret_cast<__bf16*>(arena + wave * ARENA_PER_WAVE);
#pragma unroll
    for (int mt = 0; mt < MT; ++mt)
#pragma unroll
      for (int nt = 0; nt < 4; ++nt)
#pragma unroll
        for (int i = 0; i < 8; ++i)
          q[(mt * 16 + i + rbase) * DK_ + nt * 16 + (lane & 15)] = (__bf16)qacc[mt][nt][i];
  }
  __syncthreads();

  // ================= v = Q(32x64) @ Wk[h]^T(64x56->64) =================
  v8f vacc[MT][4] = {};
  {
    const __bf16* wkh = wk_b + (size_t)wave * 64 * 64;
    const __bf16* q   = reinterpret_cast<const __bf16*>(arena + wave * ARENA_PER_WAVE);
#pragma unroll
    for (int ks = 0; ks < 2; ++ks) {
      v16bf a[MT];
#pragma unroll
      for (int mt = 0; mt < MT; ++mt) {
        u32x4* ap = reinterpret_cast<u32x4*>(&a[mt]);
        const __bf16* asrc = &q[(mt * 16 + rowA) * DK_ + ks * 32 + koffA];
        ap[0] = *reinterpret_cast<const u32x4*>(asrc);
        ap[1] = *reinterpret_cast<const u32x4*>(asrc + 16);
      }
#pragma unroll
      for (int nt = 0; nt < 4; ++nt) {
        v16bf b;
        u32x4* bp = reinterpret_cast<u32x4*>(&b);
        const __bf16* bsrc = wkh + (size_t)(nt * 16 + colB) * 64 + ks * 32 + koffB;
        bp[0] = *reinterpret_cast<const u32x4*>(bsrc);
        bp[1] = *reinterpret_cast<const u32x4*>(bsrc + 16);
#pragma unroll
        for (int mt = 0; mt < MT; ++mt)
          vacc[mt][nt] = __builtin_amdgcn_wmma_f32_16x16x32_bf16(
              false, a[mt], false, b, (short)0, vacc[mt][nt], false, false);
      }
    }
  }

  // spill v as f32 (32x56), overwriting the (now dead) Q slice of this wave
  {
    float* vv = reinterpret_cast<float*>(arena + wave * ARENA_PER_WAVE);
#pragma unroll
    for (int mt = 0; mt < MT; ++mt)
#pragma unroll
      for (int nt = 0; nt < 4; ++nt) {
        int col = nt * 16 + (lane & 15);
        if (col < D_) {
#pragma unroll
          for (int i = 0; i < 8; ++i)
            vv[(mt * 16 + i + rbase) * D_ + col] = vacc[mt][nt][i];
        }
      }
  }
  __syncthreads();

  // ================= QK (scaled) + softmax over C=8; lane == row (32 rows) =========
  const int r = lane;
  float pc[C_];
  {
    const float* vv = reinterpret_cast<const float*>(arena + wave * ARENA_PER_WAVE);
    float vrow[D_];
#pragma unroll
    for (int i = 0; i < D_ / 4; ++i)
      *reinterpret_cast<float4*>(&vrow[i * 4]) =
          reinterpret_cast<const float4*>(vv + r * D_)[i];
    float qk[C_];
#pragma unroll
    for (int c = 0; c < C_; ++c) {
      float s = 0.f;
#pragma unroll
      for (int ch = 0; ch < D_ / 8; ++ch) {
        BF8 blk;
        blk.u = *reinterpret_cast<const u32x4*>(&xbf[r * CD_ + c * D_ + ch * 8]);
#pragma unroll
        for (int k = 0; k < 8; ++k)
          s += (float)blk.h[k] * vrow[ch * 8 + k];
      }
      qk[c] = s * 0.125f;                     // 1/sqrt(DK)
    }
    float m = qk[0];
#pragma unroll
    for (int c = 1; c < C_; ++c) m = fmaxf(m, qk[c]);
    float sum = 0.f;
#pragma unroll
    for (int c = 0; c < C_; ++c) { qk[c] = __expf(qk[c] - m); sum += qk[c]; }
    float inv = 1.f / sum;
#pragma unroll
    for (int c = 0; c < C_; ++c) pc[c] = qk[c] * inv;   // stays in registers
  }
  __syncthreads();   // all waves done reading v -> arena becomes att staging

  // ================= att[r,d] = sum_c p[c]*x[r,c,d] -> stage (32 x 280) in arena ====
  {
    float* stage = reinterpret_cast<float*>(arena);
#pragma unroll
    for (int ch = 0; ch < D_ / 8; ++ch) {
      float s[8] = {};
#pragma unroll
      for (int c = 0; c < C_; ++c) {
        BF8 blk;
        blk.u = *reinterpret_cast<const u32x4*>(&xbf[r * CD_ + c * D_ + ch * 8]);
#pragma unroll
        for (int k = 0; k < 8; ++k)
          s[k] += pc[c] * (float)blk.h[k];
      }
      float* sp = &stage[r * OUTW + wave * D_ + ch * 8];
      *reinterpret_cast<float4*>(sp)     = *reinterpret_cast<float4*>(&s[0]);
      *reinterpret_cast<float4*>(sp + 4) = *reinterpret_cast<float4*>(&s[4]);
    }
  }
  __syncthreads();

  // ================= coalesced block-wide store: 32 rows x 280 f32 contiguous =======
  {
    float4* dst = reinterpret_cast<float4*>(out + (size_t)g0 * OUTW);
    const float4* src = reinterpret_cast<const float4*>(arena);
#pragma unroll 2
    for (int i = tid; i < ROWS * OUTW / 4; i += THREADS)   // 2240/160 = 14 iters
      dst[i] = src[i];
  }
}

extern "C" void kernel_launch(void* const* d_in, const int* in_sizes, int n_in,
                              void* d_out, int out_size, void* d_ws, size_t ws_size,
                              hipStream_t stream) {
  (void)in_sizes; (void)n_in; (void)out_size; (void)ws_size;
  const float* x  = (const float*)d_in[0];   // (32, 32768, 56)
  const float* Wq = (const float*)d_in[1];   // (5, 8, 56, 64)
  const float* Wk = (const float*)d_in[2];   // (5, 56, 64)
  float* out = (float*)d_out;                // (32, 4096, 280)

  const size_t wq_t_bytes = (size_t)H_ * DK_ * CD_ * sizeof(__bf16);  // 286720
  __bf16* wq_t = (__bf16*)d_ws;
  __bf16* wk_b = (__bf16*)((char*)d_ws + ((wq_t_bytes + 255) & ~(size_t)255));

  prep_wq<<<(H_ * DK_ * CD_ + 255) / 256, 256, 0, stream>>>(Wq, wq_t);
  prep_wk<<<(H_ * 64 * 64 + 255) / 256, 256, 0, stream>>>(Wk, wk_b);
  attn_fused<<<NL_ / ROWS, THREADS, 0, stream>>>(x, wq_t, wk_b, out);
}
